// MultiTaskMLP_34763465294401
// MI455X (gfx1250) — compile-verified, matrix operand order checked by reading, not run on hardware
//
#include <hip/hip_runtime.h>
#include <hip/hip_bf16.h>

typedef __attribute__((ext_vector_type(16))) _Float16 v16h;
typedef __attribute__((ext_vector_type(8)))  _Float16 v8h;
typedef __attribute__((ext_vector_type(8)))  float    v8f;

#define B_N   65536
#define D_IN  100
#define E1_N  256
#define E2_N  128
#define H_N   100
#define D1_N  64
#define D2_N  32
#define KP    128          // encoder K padded 100 -> 128
#define BN_EPS 1e-5f
#define Z2P   40           // padded f32 row stride for Z2s (32B-aligned rows)

// ---------------------------------------------------------------------------
// Unified 16-bit fragment loader (CDNA5 wave32). Source is row-major
// [outer][K] where outer = M (for A = W^T) or outer = N (for B = X^T, i.e.
// the natural row-major activation array). Lane&15 selects the row,
// lane>=16 offsets k by 8 (s<8: k=8h+s, s>=8: k=16+8h+s-8); the 16 halves are two contiguous 16-byte
// runs -> two ds_load_b128 / global_load_b128.
// ---------------------------------------------------------------------------
__device__ __forceinline__ v16h frag16(const _Float16* __restrict__ base,
                                       int strideHalves) {
  const int lane = threadIdx.x & 31;
  const _Float16* p = base + (lane & 15) * strideHalves + ((lane >> 4) << 3);
  v8h lo = *(const v8h*)(p);
  v8h hi = *(const v8h*)(p + 16);
  return __builtin_shufflevector(lo, hi, 0, 1, 2, 3, 4, 5, 6, 7,
                                 8, 9, 10, 11, 12, 13, 14, 15);
}

__device__ __forceinline__ v8h epilogue_pack(v8f acc, v8f sA, v8f sC) {
  v8h oh;
#pragma unroll
  for (int v = 0; v < 8; ++v) {
    float z = acc[v] * sA[v] + sC[v];
    oh[v] = (_Float16)(z > 0.f ? z : 0.f);
  }
  return oh;
}

// ---------------------------------------------------------------------------
// prep: fold BN(running stats)+bias into per-feature (A, C); weights -> f16,
// stored TRANSPOSED ([out_features][K]) so LDS staging + frags are contiguous
// ---------------------------------------------------------------------------
__global__ void __launch_bounds__(256)
prep_kernel(const float* __restrict__ eW1, const float* __restrict__ eb1,
            const float* __restrict__ eg1, const float* __restrict__ ebt1,
            const float* __restrict__ em1, const float* __restrict__ ev1,
            const float* __restrict__ eW2, const float* __restrict__ eb2,
            const float* __restrict__ eg2, const float* __restrict__ ebt2,
            const float* __restrict__ em2, const float* __restrict__ ev2,
            const float* __restrict__ dW1, const float* __restrict__ db1,
            const float* __restrict__ dg1, const float* __restrict__ dbt1,
            const float* __restrict__ dm1, const float* __restrict__ dv1,
            const float* __restrict__ dW2, const float* __restrict__ db2,
            const float* __restrict__ dg2, const float* __restrict__ dbt2,
            const float* __restrict__ dm2, const float* __restrict__ dv2,
            _Float16* __restrict__ eW1hT, _Float16* __restrict__ eW2hT,
            float* __restrict__ A1, float* __restrict__ C1,
            float* __restrict__ A2, float* __restrict__ C2,
            _Float16* __restrict__ dW1hT, _Float16* __restrict__ dW2hT,
            float* __restrict__ dA1, float* __restrict__ dC1,
            float* __restrict__ dA2, float* __restrict__ dC2,
            int* __restrict__ counts) {
  const int i = blockIdx.x * blockDim.x + threadIdx.x;
  if (i < E1_N * KP) {                       // eW1^T [E1][KP], K-padded
    const int n = i / KP, k = i % KP;
    eW1hT[i] = (k < D_IN) ? (_Float16)eW1[k * E1_N + n] : (_Float16)0.f;
  }
  if (i < E2_N * E1_N) {                     // eW2^T [E2][E1]
    const int n = i / E1_N, k = i % E1_N;
    eW2hT[i] = (_Float16)eW2[k * E2_N + n];
  }
  if (i < E1_N) {
    const float a = eg1[i] * rsqrtf(ev1[i] + BN_EPS);
    A1[i] = a;  C1[i] = (eb1[i] - em1[i]) * a + ebt1[i];
  }
  if (i < E2_N) {
    const float a = eg2[i] * rsqrtf(ev2[i] + BN_EPS);
    A2[i] = a;  C2[i] = (eb2[i] - em2[i]) * a + ebt2[i];
  }
  if (i < H_N * D1_N * E2_N) {               // dW1^T [H][D1][E2]
    const int h = i / (D1_N * E2_N), r = i % (D1_N * E2_N);
    const int o = r / E2_N, k = r % E2_N;
    dW1hT[i] = (_Float16)dW1[(size_t)h * E2_N * D1_N + k * D1_N + o];
  }
  if (i < H_N * D2_N * D1_N) {               // dW2^T [H][D2][D1]
    const int h = i / (D2_N * D1_N), r = i % (D2_N * D1_N);
    const int o = r / D1_N, k = r % D1_N;
    dW2hT[i] = (_Float16)dW2[(size_t)h * D1_N * D2_N + k * D2_N + o];
  }
  if (i < H_N * D1_N) {
    const float a = dg1[i] * rsqrtf(dv1[i] + BN_EPS);
    dA1[i] = a;  dC1[i] = (db1[i] - dm1[i]) * a + dbt1[i];
  }
  if (i < H_N * D2_N) {
    const float a = dg2[i] * rsqrtf(dv2[i] + BN_EPS);
    dA2[i] = a;  dC2[i] = (db2[i] - dm2[i]) * a + dbt2[i];
  }
  if (i < H_N) counts[i] = 0;
}

// ---------------------------------------------------------------------------
// expert grouping: count -> exclusive scan -> scatter sample indices
// ---------------------------------------------------------------------------
__global__ void __launch_bounds__(256)
count_kernel(const int* __restrict__ ids, int* __restrict__ counts) {
  const int i = blockIdx.x * blockDim.x + threadIdx.x;
  if (i < B_N) atomicAdd(&counts[ids[i]], 1);
}

__global__ void __launch_bounds__(32)
scan_kernel(const int* __restrict__ counts, int* __restrict__ offsets,
            int* __restrict__ cursor) {
  if (threadIdx.x == 0 && blockIdx.x == 0) {
    int acc = 0;
    for (int e = 0; e < H_N; ++e) { offsets[e] = acc; cursor[e] = acc; acc += counts[e]; }
    offsets[H_N] = acc;
  }
}

__global__ void __launch_bounds__(256)
scatter_kernel(const int* __restrict__ ids, int* __restrict__ cursor,
               int* __restrict__ order) {
  const int i = blockIdx.x * blockDim.x + threadIdx.x;
  if (i < B_N) {
    const int pos = atomicAdd(&cursor[ids[i]], 1);
    order[pos] = i;
  }
}

// ---------------------------------------------------------------------------
// encoder: 32 samples/block, 4 waves. Computes h^T = W^T x^T per tile:
// A-frag from transposed weight panel, B-frag straight from row-major
// activations, D epilogue stores 8 contiguous features per lane (b128).
// ---------------------------------------------------------------------------
__global__ void __launch_bounds__(128)
encoder_kernel(const float* __restrict__ x,
               const _Float16* __restrict__ eW1hT,  // [E1][KP]
               const _Float16* __restrict__ eW2hT,  // [E2][E1]
               const float* __restrict__ A1, const float* __restrict__ C1,
               const float* __restrict__ A2, const float* __restrict__ C2,
               _Float16* __restrict__ h2h) {
  __shared__ __align__(16) _Float16 Xs[32 * KP];      //  8 KB x tile (f16)
  __shared__ __align__(16) _Float16 Wb[128 * 128];    // 32 KB weight panel
  __shared__ __align__(16) _Float16 h1s[32 * E1_N];   // 16 KB hidden1 tile

  const int tid  = threadIdx.x;
  const int wave = tid >> 5;
  const int lane = tid & 31;
  const int hi8  = (lane >> 4) << 3;
  const int row0 = blockIdx.x * 32;

  // load + convert x tile (zero-pad K 100->128)
  for (int i = tid; i < 32 * KP; i += 128) {
    const int r = i >> 7, k = i & 127;
    const float v = (k < D_IN) ? x[(size_t)(row0 + r) * D_IN + k] : 0.f;
    Xs[i] = (_Float16)v;
  }
  __syncthreads();

  // ---- GEMM1: 256 output features in two panels of 128 ----
  for (int panel = 0; panel < 2; ++panel) {
    {  // contiguous uint4 staging of W1^T rows [panel*128, +128)
      const uint4* src = (const uint4*)(eW1hT + (size_t)panel * 128 * KP);
      uint4* dst = (uint4*)Wb;
      for (int i = tid; i < 128 * KP / 8; i += 128) dst[i] = src[i];
    }
    __syncthreads();
#pragma unroll
    for (int f2 = 0; f2 < 2; ++f2) {
      const int ft = wave * 2 + f2;
#pragma unroll
      for (int st = 0; st < 2; ++st) {
        v8f acc = {0.f, 0.f, 0.f, 0.f, 0.f, 0.f, 0.f, 0.f};
#pragma unroll
        for (int ks = 0; ks < 4; ++ks) {
          v16h a = frag16(&Wb[(ft * 16) * KP + ks * 32], KP);
          v16h b = frag16(&Xs[(st * 16) * KP + ks * 32], KP);
          acc = __builtin_amdgcn_wmma_f32_16x16x32_f16(false, a, false, b,
                                                       (short)0, acc, false, false);
        }
        const int f0 = panel * 128 + ft * 16 + hi8;   // feature base (x8)
        const int s  = st * 16 + (lane & 15);         // sample row
        v8f sA = *(const v8f*)(A1 + f0);
        v8f sC = *(const v8f*)(C1 + f0);
        *(v8h*)(&h1s[s * E1_N + f0]) = epilogue_pack(acc, sA, sC);
      }
    }
    __syncthreads();
  }

  // ---- GEMM2: K=256 staged as two panels, accumulate in registers ----
  v8f zero = {0.f, 0.f, 0.f, 0.f, 0.f, 0.f, 0.f, 0.f};
  v8f acc2[2][2] = {{zero, zero}, {zero, zero}};
  for (int kh = 0; kh < 2; ++kh) {
    for (int i = tid; i < 128 * 128 / 8; i += 128) {   // Wb[f][k] panel
      const int f = i >> 4, q = i & 15;
      ((uint4*)Wb)[i] =
          ((const uint4*)(eW2hT + (size_t)f * E1_N + kh * 128))[q];
    }
    __syncthreads();
#pragma unroll
    for (int f2 = 0; f2 < 2; ++f2) {
      const int ft = wave * 2 + f2;
#pragma unroll
      for (int st = 0; st < 2; ++st) {
#pragma unroll
        for (int ks = 0; ks < 4; ++ks) {
          v16h a = frag16(&Wb[(ft * 16) * 128 + ks * 32], 128);
          v16h b = frag16(&h1s[(st * 16) * E1_N + kh * 128 + ks * 32], E1_N);
          acc2[f2][st] = __builtin_amdgcn_wmma_f32_16x16x32_f16(
              false, a, false, b, (short)0, acc2[f2][st], false, false);
        }
      }
    }
    __syncthreads();
  }

  // epilogue -> global h2 (f16, b128 stores)
#pragma unroll
  for (int f2 = 0; f2 < 2; ++f2) {
    const int ft = wave * 2 + f2;
#pragma unroll
    for (int st = 0; st < 2; ++st) {
      const int f0 = ft * 16 + hi8;
      const int s  = row0 + st * 16 + (lane & 15);
      v8f sA = *(const v8f*)(A2 + f0);
      v8f sC = *(const v8f*)(C2 + f0);
      *(v8h*)(&h2h[(size_t)s * E2_N + f0]) = epilogue_pack(acc2[f2][st], sA, sC);
    }
  }
}

// ---------------------------------------------------------------------------
// decoder: one block per (expert, chunk-stripe); expert weights (transposed)
// staged in LDS once; 64 gathered rows per chunk; two WMMA GEMMs + final dot.
// ---------------------------------------------------------------------------
__global__ void __launch_bounds__(128)
decoder_kernel(const _Float16* __restrict__ h2h,
               const _Float16* __restrict__ dW1hT,   // [H][D1][E2]
               const _Float16* __restrict__ dW2hT,   // [H][D2][D1]
               const float* __restrict__ dA1, const float* __restrict__ dC1,
               const float* __restrict__ dA2, const float* __restrict__ dC2,
               const float* __restrict__ dW3, const float* __restrict__ db3,
               const int* __restrict__ offsets,
               const int* __restrict__ order,
               float* __restrict__ out) {
  __shared__ __align__(16) _Float16 As[64 * E2_N];        // 16 KB gathered h2
  __shared__ __align__(16) _Float16 W1s[D1_N * E2_N];     // 16 KB W1^T
  __shared__ __align__(16) _Float16 Z1s[64 * D1_N];       //  8 KB hidden (f16)
  __shared__ __align__(16) _Float16 W2s[D2_N * D1_N];     //  4 KB W2^T
  __shared__ __align__(32) float    Z2s[64 * Z2P];        // 10 KB hidden2 f32
  __shared__ int   rowIdx[64];
  __shared__ __align__(32) float sA1[D1_N], sC1[D1_N];
  __shared__ __align__(32) float sA2[D2_N], sC2[D2_N], sW3[D2_N];

  const int e    = blockIdx.x;
  const int tid  = threadIdx.x;
  const int wave = tid >> 5;
  const int lane = tid & 31;
  const int hi8  = (lane >> 4) << 3;
  const int beg  = offsets[e];
  const int cnt  = offsets[e + 1] - beg;

  // stage this expert's parameters once (contiguous uint4 copies)
  {
    const uint4* s1 = (const uint4*)(dW1hT + (size_t)e * D1_N * E2_N);
    for (int i = tid; i < D1_N * E2_N / 8; i += 128) ((uint4*)W1s)[i] = s1[i];
    const uint4* s2 = (const uint4*)(dW2hT + (size_t)e * D2_N * D1_N);
    for (int i = tid; i < D2_N * D1_N / 8; i += 128) ((uint4*)W2s)[i] = s2[i];
  }
  if (tid < D1_N) { sA1[tid] = dA1[e * D1_N + tid]; sC1[tid] = dC1[e * D1_N + tid]; }
  if (tid < D2_N) {
    sA2[tid] = dA2[e * D2_N + tid]; sC2[tid] = dC2[e * D2_N + tid];
    sW3[tid] = dW3[e * D2_N + tid];
  }
  const float bias3 = db3[e];

  for (int chunk = blockIdx.y; chunk * 64 < cnt; chunk += gridDim.y) {
    const int base = chunk * 64;
    if (tid < 64) rowIdx[tid] = (base + tid < cnt) ? order[beg + base + tid] : -1;
    __syncthreads();

    // gather 64 rows of h2 (16B vector copies; row = 16 uint4)
    for (int i = tid; i < 64 * 16; i += 128) {
      const int r = i >> 4, q = i & 15;
      const int src = rowIdx[r];
      uint4 v = {0u, 0u, 0u, 0u};
      if (src >= 0) v = ((const uint4*)(h2h + (size_t)src * E2_N))[q];
      ((uint4*)As)[i] = v;
    }
    __syncthreads();

    // GEMM d1: z1^T = W1^T h2^T  ([64f] x [K=128] x [64s])
#pragma unroll
    for (int st = 0; st < 4; ++st) {
      v8f acc = {0.f, 0.f, 0.f, 0.f, 0.f, 0.f, 0.f, 0.f};
#pragma unroll
      for (int ks = 0; ks < 4; ++ks) {
        v16h a = frag16(&W1s[(wave * 16) * E2_N + ks * 32], E2_N);
        v16h b = frag16(&As[(st * 16) * E2_N + ks * 32], E2_N);
        acc = __builtin_amdgcn_wmma_f32_16x16x32_f16(false, a, false, b,
                                                     (short)0, acc, false, false);
      }
      const int f0 = wave * 16 + hi8;
      const int s  = st * 16 + (lane & 15);
      v8f sA = *(const v8f*)(sA1 + f0);
      v8f sC = *(const v8f*)(sC1 + f0);
      *(v8h*)(&Z1s[s * D1_N + f0]) = epilogue_pack(acc, sA, sC);
    }
    __syncthreads();

    // GEMM d2: z2^T = W2^T z1^T  ([32f] x [K=64] x [64s])
    {
      const int ft = wave >> 1;
      const int s0 = (wave & 1) * 2;
#pragma unroll
      for (int s2i = 0; s2i < 2; ++s2i) {
        const int st = s0 + s2i;
        v8f acc = {0.f, 0.f, 0.f, 0.f, 0.f, 0.f, 0.f, 0.f};
#pragma unroll
        for (int ks = 0; ks < 2; ++ks) {
          v16h a = frag16(&W2s[(ft * 16) * D1_N + ks * 32], D1_N);
          v16h b = frag16(&Z1s[(st * 16) * D1_N + ks * 32], D1_N);
          acc = __builtin_amdgcn_wmma_f32_16x16x32_f16(false, a, false, b,
                                                       (short)0, acc, false, false);
        }
        const int f0 = ft * 16 + hi8;
        const int s  = st * 16 + (lane & 15);
        v8f sA = *(const v8f*)(sA2 + f0);
        v8f sC = *(const v8f*)(sC2 + f0);
        v8f zr;
#pragma unroll
        for (int v = 0; v < 8; ++v) {
          float z = acc[v] * sA[v] + sC[v];
          zr[v] = (z > 0.f ? z : 0.f);
        }
        *(v8f*)(&Z2s[s * Z2P + f0]) = zr;
      }
    }
    __syncthreads();

    // final layer: logits[b] = z2 . w3 + b3, scattered to original order
    if (tid < 64) {
      const int src = rowIdx[tid];
      if (src >= 0) {
        float s = bias3;
#pragma unroll
        for (int c = 0; c < D2_N; ++c) s += Z2s[tid * Z2P + c] * sW3[c];
        out[src] = s;
      }
    }
    __syncthreads();
  }
}

// ---------------------------------------------------------------------------
extern "C" void kernel_launch(void* const* d_in, const int* in_sizes, int n_in,
                              void* d_out, int out_size, void* d_ws, size_t ws_size,
                              hipStream_t stream) {
  const float* x    = (const float*)d_in[0];
  const float* eW1  = (const float*)d_in[1];
  const float* eb1  = (const float*)d_in[2];
  const float* eg1  = (const float*)d_in[3];
  const float* ebt1 = (const float*)d_in[4];
  const float* em1  = (const float*)d_in[5];
  const float* ev1  = (const float*)d_in[6];
  const float* eW2  = (const float*)d_in[7];
  const float* eb2  = (const float*)d_in[8];
  const float* eg2  = (const float*)d_in[9];
  const float* ebt2 = (const float*)d_in[10];
  const float* em2  = (const float*)d_in[11];
  const float* ev2  = (const float*)d_in[12];
  const float* dW1  = (const float*)d_in[13];
  const float* db1  = (const float*)d_in[14];
  const float* dg1  = (const float*)d_in[15];
  const float* dbt1 = (const float*)d_in[16];
  const float* dm1  = (const float*)d_in[17];
  const float* dv1  = (const float*)d_in[18];
  const float* dW2  = (const float*)d_in[19];
  const float* db2  = (const float*)d_in[20];
  const float* dg2  = (const float*)d_in[21];
  const float* dbt2 = (const float*)d_in[22];
  const float* dm2  = (const float*)d_in[23];
  const float* dv2  = (const float*)d_in[24];
  const float* dW3  = (const float*)d_in[25];
  const float* db3  = (const float*)d_in[26];
  const int*   ids  = (const int*)d_in[27];
  float* out = (float*)d_out;

  // workspace carve-up (256B aligned regions)
  char* ws = (char*)d_ws;
  size_t o = 0;
  auto alloc = [&](size_t bytes) -> void* {
    void* p = ws + o;
    o = (o + bytes + 255) & ~(size_t)255;
    return p;
  };
  _Float16* eW1hT = (_Float16*)alloc((size_t)E1_N * KP * 2);
  _Float16* eW2hT = (_Float16*)alloc((size_t)E2_N * E1_N * 2);
  float* A1 = (float*)alloc(E1_N * 4);
  float* C1 = (float*)alloc(E1_N * 4);
  float* A2 = (float*)alloc(E2_N * 4);
  float* C2 = (float*)alloc(E2_N * 4);
  _Float16* dW1hT = (_Float16*)alloc((size_t)H_N * D1_N * E2_N * 2);
  _Float16* dW2hT = (_Float16*)alloc((size_t)H_N * D2_N * D1_N * 2);
  float* dA1 = (float*)alloc((size_t)H_N * D1_N * 4);
  float* dC1 = (float*)alloc((size_t)H_N * D1_N * 4);
  float* dA2 = (float*)alloc((size_t)H_N * D2_N * 4);
  float* dC2 = (float*)alloc((size_t)H_N * D2_N * 4);
  int* counts  = (int*)alloc(H_N * 4);
  int* offsets = (int*)alloc((H_N + 1) * 4);
  int* cursor  = (int*)alloc(H_N * 4);
  int* order   = (int*)alloc((size_t)B_N * 4);
  _Float16* h2h = (_Float16*)alloc((size_t)B_N * E2_N * 2);
  (void)ws_size; (void)in_sizes; (void)n_in; (void)out_size;

  const int prepN = H_N * D1_N * E2_N;  // largest prep array (819200)
  prep_kernel<<<(prepN + 255) / 256, 256, 0, stream>>>(
      eW1, eb1, eg1, ebt1, em1, ev1, eW2, eb2, eg2, ebt2, em2, ev2,
      dW1, db1, dg1, dbt1, dm1, dv1, dW2, db2, dg2, dbt2, dm2, dv2,
      eW1hT, eW2hT, A1, C1, A2, C2, dW1hT, dW2hT, dA1, dC1, dA2, dC2, counts);

  count_kernel<<<(B_N + 255) / 256, 256, 0, stream>>>(ids, counts);
  scan_kernel<<<1, 32, 0, stream>>>(counts, offsets, cursor);
  scatter_kernel<<<(B_N + 255) / 256, 256, 0, stream>>>(ids, cursor, order);

  encoder_kernel<<<B_N / 32, 128, 0, stream>>>(x, eW1hT, eW2hT, A1, C1, A2, C2, h2h);

  decoder_kernel<<<dim3(H_N, 8), 128, 0, stream>>>(
      h2h, dW1hT, dW2hT, dA1, dC1, dA2, dC2, dW3, db3, offsets, order, out);
}
